// HyperbolicHPE_1297080123954
// MI455X (gfx1250) — compile-verified, matrix operand order checked by reading, not run on hardware
//
#include <hip/hip_runtime.h>
#include <hip/hip_bf16.h>
#include <math.h>

// ---------------------------------------------------------------------------
// HyperbolicHPE forward for gfx1250 (MI455X).
// GEMMs use v_wmma_f32_16x16x32_f16 with LDS double-buffering, b128 global
// staging, and 32x32-per-wave register tiling (2 LDS b128 loads per WMMA).
// Attention pieces (J=17 spatial, windowed temporal) are latency-bound
// bespoke wave32 kernels with shfl reductions / online softmax.
// ---------------------------------------------------------------------------

typedef __attribute__((ext_vector_type(16))) _Float16 v16h;
typedef __attribute__((ext_vector_type(8)))  _Float16 v8h;
typedef __attribute__((ext_vector_type(4)))  _Float16 v4h;
typedef __attribute__((ext_vector_type(8)))  float    v8f;
typedef __attribute__((ext_vector_type(4)))  float    v4f;

#define TILE_M 128
#define TILE_N 64
#define TILE_K 64
#define LDS_STRIDE (TILE_K + 8)   // 72 halves = 144B rows; keeps 16B alignment

// ---------------------------------------------------------------------------
// Stage one K-tile of A (128x64) and B (64x64) into registers
// (12 x global_load_b128 per thread). M-edge via index clamp + select
// (no divergent exec masking).
// ---------------------------------------------------------------------------
__device__ __forceinline__ void load_tile_regs(
    const float* __restrict__ A, int lda,
    const float* __restrict__ B, int ldb,
    int tileM, int tileN, int k0, int M, bool edge, int tid,
    v4f (&areg)[8], v4f (&breg)[4])
{
#pragma unroll
  for (int i = 0; i < 8; ++i) {
    int e  = i * 256 + tid;
    int r  = e >> 4;           // 0..127
    int c4 = (e & 15) << 2;    // 0,4,..,60
    int gr = tileM + r;
    if (edge) {
      int grc = gr < M ? gr : (M - 1);
      v4f v = *(const v4f*)&A[(size_t)grc * lda + k0 + c4];
      if (gr >= M) v = (v4f){0.f, 0.f, 0.f, 0.f};
      areg[i] = v;
    } else {
      areg[i] = *(const v4f*)&A[(size_t)gr * lda + k0 + c4];
    }
  }
#pragma unroll
  for (int i = 0; i < 4; ++i) {
    int e  = i * 256 + tid;
    int r  = e >> 4;           // k row 0..63
    int c4 = (e & 15) << 2;    // n 0..60
    breg[i] = *(const v4f*)&B[(size_t)(k0 + r) * ldb + tileN + c4];
  }
}

// Convert staged registers to f16 and write into an LDS buffer.
// A: row-major [m][k] (one ds_store_b64 per 4 elems).
// B: transposed [n][k] (4 x ds_store_b16).
__device__ __forceinline__ void store_tile_lds(
    _Float16 (*lA)[LDS_STRIDE], _Float16 (*lB)[LDS_STRIDE],
    int tid, const v4f (&areg)[8], const v4f (&breg)[4])
{
#pragma unroll
  for (int i = 0; i < 8; ++i) {
    int e  = i * 256 + tid;
    int r  = e >> 4;
    int c4 = (e & 15) << 2;
    v4h h;
#pragma unroll
    for (int u = 0; u < 4; ++u) h[u] = (_Float16)areg[i][u];
    *(v4h*)&lA[r][c4] = h;
  }
#pragma unroll
  for (int i = 0; i < 4; ++i) {
    int e  = i * 256 + tid;
    int r  = e >> 4;
    int c4 = (e & 15) << 2;
#pragma unroll
    for (int u = 0; u < 4; ++u) lB[c4 + u][r] = (_Float16)breg[i][u];
  }
}

// ---------------------------------------------------------------------------
// Tiled WMMA GEMM: C[M,N] = A[M,K] @ B[K,N] (+bias[N]) (+residual) (gelu?)
// flags: bit0 = gelu, bit1 = add residual.
// Workgroup tile 128x64, 8 waves arranged 4(M) x 2(N); each wave owns a
// 32x32 C slab (2x2 16x16 WMMA tiles) -> 8 WMMAs per 64-deep K step with
// each A/B fragment reused twice (2 ds_load_b128 per WMMA).
// Double-buffered LDS: next tile's global loads are in flight during compute.
// Requires: K % 64 == 0, N % 64 == 0 (true for all uses here).
// ---------------------------------------------------------------------------
__global__ __launch_bounds__(256) void gemm_f16wmma(
    const float* __restrict__ A, int lda,
    const float* __restrict__ B, int ldb,
    const float* __restrict__ bias,
    const float* __restrict__ residual,
    float* __restrict__ C, int ldc,
    int M, int N, int K, int flags)
{
  __shared__ _Float16 lA[2][TILE_M][LDS_STRIDE];
  __shared__ _Float16 lB[2][TILE_N][LDS_STRIDE];

  const int tid   = threadIdx.x;
  const int tileM = blockIdx.x * TILE_M;
  const int tileN = blockIdx.y * TILE_N;
  const int wave  = tid >> 5;
  const int lane  = tid & 31;
  const int wm    = wave & 3;     // M group (32 rows) 0..3
  const int wn    = wave >> 2;    // N group (32 cols) 0..1
  const int half  = lane >> 4;    // 0: lanes 0-15, 1: lanes 16-31
  const int l16   = lane & 15;
  const bool edge = (tileM + TILE_M) > M;   // uniform per block

  v8f acc[2][2] = {};
  v4f areg[8];
  v4f breg[4];

  const int ktiles = K / TILE_K;

  load_tile_regs(A, lda, B, ldb, tileM, tileN, 0, M, edge, tid, areg, breg);
  store_tile_lds(lA[0], lB[0], tid, areg, breg);
  __syncthreads();

  for (int kt = 0; kt < ktiles; ++kt) {
    const int cur = kt & 1;
    const bool more = (kt + 1) < ktiles;
    if (more)
      load_tile_regs(A, lda, B, ldb, tileM, tileN, (kt + 1) * TILE_K,
                     M, edge, tid, areg, breg);

#pragma unroll
    for (int kh = 0; kh < 2; ++kh) {
      // A fragments: rows wm*32 + ms*16 + l16
      v16h af[2];
#pragma unroll
      for (int ms = 0; ms < 2; ++ms) {
        const _Float16* pa = &lA[cur][wm * 32 + ms * 16 + l16][kh * 32 + half * 8];
        v8h alo = *(const v8h*)pa;          // K 0..7 / 8..15 (of this half)
        v8h ahi = *(const v8h*)(pa + 16);   // K 16..23 / 24..31
#pragma unroll
        for (int i = 0; i < 8; ++i) { af[ms][i] = alo[i]; af[ms][8 + i] = ahi[i]; }
      }
      // B fragments: cols wn*32 + ns*16 + l16
      v16h bf[2];
#pragma unroll
      for (int ns = 0; ns < 2; ++ns) {
        const _Float16* pb = &lB[cur][wn * 32 + ns * 16 + l16][kh * 32 + half * 8];
        v8h blo = *(const v8h*)pb;
        v8h bhi = *(const v8h*)(pb + 16);
#pragma unroll
        for (int i = 0; i < 8; ++i) { bf[ns][i] = blo[i]; bf[ns][8 + i] = bhi[i]; }
      }
#pragma unroll
      for (int ms = 0; ms < 2; ++ms)
#pragma unroll
        for (int ns = 0; ns < 2; ++ns)
          acc[ms][ns] = __builtin_amdgcn_wmma_f32_16x16x32_f16(
              false, af[ms], false, bf[ns], (short)0, acc[ms][ns], false, false);
    }

    if (more)
      store_tile_lds(lA[1 - cur], lB[1 - cur], tid, areg, breg);
    __syncthreads();
  }

  // ---- epilogue: C layout VGPR r -> M=r (lanes 0-15) / M=r+8 (16-31) ----
  const bool do_gelu = (flags & 1) != 0;
  const bool do_res  = (flags & 2) != 0;
#pragma unroll
  for (int ms = 0; ms < 2; ++ms) {
#pragma unroll
    for (int ns = 0; ns < 2; ++ns) {
      const int col = tileN + wn * 32 + ns * 16 + l16;
#pragma unroll
      for (int r = 0; r < 8; ++r) {
        int row = tileM + wm * 32 + ms * 16 + half * 8 + r;
        if (row < M) {
          float v = acc[ms][ns][r];
          if (bias)    v += bias[col];
          if (do_res)  v += residual[(size_t)row * ldc + col];
          if (do_gelu) v = 0.5f * v * (1.0f + erff(v * 0.70710678118f));
          C[(size_t)row * ldc + col] = v;
        }
      }
    }
  }
}

// ---------------------------------------------------------------------------
// LayerNorm over last dim (D=512), one block (256 thr) per token.
// ---------------------------------------------------------------------------
__global__ __launch_bounds__(256) void layernorm_k(
    const float* __restrict__ x, const float* __restrict__ g,
    const float* __restrict__ b, float* __restrict__ y)
{
  __shared__ float rs[256], rq[256];
  const int t = blockIdx.x, tid = threadIdx.x;
  const float* xr = x + (size_t)t * 512;
  float v0 = xr[tid], v1 = xr[tid + 256];
  rs[tid] = v0 + v1;
  rq[tid] = v0 * v0 + v1 * v1;
  __syncthreads();
  for (int s = 128; s > 0; s >>= 1) {
    if (tid < s) { rs[tid] += rs[tid + s]; rq[tid] += rq[tid + s]; }
    __syncthreads();
  }
  const float mean = rs[0] * (1.0f / 512.0f);
  const float var  = rq[0] * (1.0f / 512.0f) - mean * mean;
  const float rstd = rsqrtf(var + 1e-5f);
  float* yr = y + (size_t)t * 512;
  yr[tid]       = (v0 - mean) * rstd * g[tid]       + b[tid];
  yr[tid + 256] = (v1 - mean) * rstd * g[tid + 256] + b[tid + 256];
}

// ---------------------------------------------------------------------------
// Embedding: h = x@Wp + bp (K=3); h_x = h * min(1, MAX/||h||) + joint_embed
// (exp_map0 then log_map0 reduces to a norm clip); h_v = xv[:,:2]@Wvel + bvel.
// ---------------------------------------------------------------------------
__global__ __launch_bounds__(256) void embed_k(
    const float* __restrict__ x, const float* __restrict__ xv,
    const float* __restrict__ Wp, const float* __restrict__ bp,
    const float* __restrict__ Wvel, const float* __restrict__ bvel,
    const float* __restrict__ jemb,
    float* __restrict__ hx, float* __restrict__ hv)
{
  __shared__ float red[256];
  const int t = blockIdx.x, tid = threadIdx.x;
  const int j = t % 17;
  const float x0 = x[t * 3 + 0], x1 = x[t * 3 + 1], x2 = x[t * 3 + 2];
  const float u0 = xv[t * 3 + 0], u1 = xv[t * 3 + 1];

  float h[2]; float ss = 0.0f;
#pragma unroll
  for (int i = 0; i < 2; ++i) {
    int d = tid + i * 256;
    h[i] = x0 * Wp[d] + x1 * Wp[512 + d] + x2 * Wp[1024 + d] + bp[d];
    ss += h[i] * h[i];
  }
  red[tid] = ss; __syncthreads();
  for (int s = 128; s > 0; s >>= 1) {
    if (tid < s) red[tid] += red[tid + s];
    __syncthreads();
  }
  const float n2 = fmaxf(red[0], 1e-7f);
  const float n  = sqrtf(n2);
  const float scale = fminf(1.0f, 15.0f / n);
#pragma unroll
  for (int i = 0; i < 2; ++i) {
    int d = tid + i * 256;
    hx[(size_t)t * 512 + d] = h[i] * scale + jemb[j * 512 + d];
    hv[(size_t)t * 512 + d] = u0 * Wvel[d] + u1 * Wvel[512 + d] + bvel[d];
  }
}

// ---------------------------------------------------------------------------
// Spatial attention, one block per (token-group n, head h). J=17, dh=64.
// Hyperbolic distance: inner = s_i*s_j*(q.k) - c_i*c_j with
//   s = sinh(min(n,MAX))/n_orig, c = cosh(min(n,MAX)).
// ---------------------------------------------------------------------------
__global__ __launch_bounds__(256) void spatial_attn_k(
    const float* __restrict__ q, const float* __restrict__ k,
    const float* __restrict__ v, const float* __restrict__ Atopo,
    const float* __restrict__ gamma, float* __restrict__ z)
{
  __shared__ float sq[17][64], sk[17][64], sv[17][64];
  __shared__ float qs[17], qc[17], ks[17], kc[17];
  __shared__ float sc[17][18];
  const int n = blockIdx.x, h = blockIdx.y, tid = threadIdx.x;

  for (int e = tid; e < 17 * 64; e += 256) {
    int i = e >> 6, d = e & 63;
    size_t off = ((size_t)(n * 17 + i)) * 512 + h * 64 + d;
    sq[i][d] = q[off]; sk[i][d] = k[off]; sv[i][d] = v[off];
  }
  __syncthreads();

  if (tid < 34) {
    int i = tid % 17; bool isq = tid < 17;
    const float* row = isq ? sq[i] : sk[i];
    float ss = 0.0f;
    for (int d = 0; d < 64; ++d) ss += row[d] * row[d];
    float n2 = fmaxf(ss, 1e-7f), no = sqrtf(n2);
    float nc = fminf(no, 15.0f);
    float s = sinhf(nc) / no, c = coshf(nc);
    if (isq) { qs[i] = s; qc[i] = c; } else { ks[i] = s; kc[i] = c; }
  }
  __syncthreads();

  const float g = gamma[h];
  if (tid < 289) {
    int i = tid / 17, jj = tid % 17;
    float dot = 0.0f;
    for (int d = 0; d < 64; ++d) dot += sq[i][d] * sk[jj][d];
    float inner = qs[i] * ks[jj] * dot - qc[i] * kc[jj];
    float xx = fmaxf(-inner, 1.0f + 1e-7f);
    float dist = logf(xx + sqrtf(xx * xx - 1.0f));   // arccosh
    sc[i][jj] = -dist * 0.125f + g * Atopo[i * 17 + jj];
  }
  __syncthreads();

  if (tid < 17) {
    float m = -1e30f;
    for (int jj = 0; jj < 17; ++jj) m = fmaxf(m, sc[tid][jj]);
    float s = 0.0f;
    for (int jj = 0; jj < 17; ++jj) { float e = __expf(sc[tid][jj] - m); sc[tid][jj] = e; s += e; }
    float inv = 1.0f / s;
    for (int jj = 0; jj < 17; ++jj) sc[tid][jj] *= inv;
  }
  __syncthreads();

  for (int e = tid; e < 17 * 64; e += 256) {
    int i = e >> 6, d = e & 63;
    float acc = 0.0f;
    for (int jj = 0; jj < 17; ++jj) acc += sc[i][jj] * sv[jj][d];
    z[((size_t)(n * 17 + i)) * 512 + h * 64 + d] = acc;
  }
}

__device__ __forceinline__ float wsum32(float v) {
#pragma unroll
  for (int off = 16; off > 0; off >>= 1) v += __shfl_xor(v, off, 32);
  return v;
}

// ---------------------------------------------------------------------------
// Temporal windowed attention. One wave per query; lane l covers dims l, l+32.
// Online softmax over window [i-R, i+R]. Grid: x = query chunks of 8,
// y = ((b*J + j)*8 + h).
// ---------------------------------------------------------------------------
__global__ __launch_bounds__(256) void temporal_attn_k(
    const float* __restrict__ q, const float* __restrict__ k,
    const float* __restrict__ v, float* __restrict__ z,
    int B, int T, int J, int R)
{
  const int wid = threadIdx.x >> 5, lane = threadIdx.x & 31;
  const int i = blockIdx.x * 8 + wid;
  const int y = blockIdx.y;
  const int h = y & 7, bj = y >> 3;
  const int j = bj % J, b = bj / J;
  if (i >= T) return;

  const size_t qoff = ((size_t)(b * T + i) * J + j) * 512 + h * 64 + lane;
  const float qa = q[qoff], qb = q[qoff + 32];

  float n2q = fmaxf(wsum32(qa * qa + qb * qb), 1e-7f);
  float noq = sqrtf(n2q), ncq = fminf(noq, 15.0f);
  float si = sinhf(ncq) / noq, ci = coshf(ncq);

  const int lo = max(i - R, 0), hi = min(i + R, T - 1);
  float m = -1e30f, Z = 0.0f, za = 0.0f, zb = 0.0f;
  for (int t = lo; t <= hi; ++t) {
    size_t off = ((size_t)(b * T + t) * J + j) * 512 + h * 64 + lane;
    float ka = k[off], kb = k[off + 32];
    float dot = wsum32(qa * ka + qb * kb);
    float kk  = fmaxf(wsum32(ka * ka + kb * kb), 1e-7f);
    float nok = sqrtf(kk), nck = fminf(nok, 15.0f);
    float sj = sinhf(nck) / nok, cj = coshf(nck);
    float inner = si * sj * dot - ci * cj;
    float xx = fmaxf(-inner, 1.0f + 1e-7f);
    float s = -logf(xx + sqrtf(xx * xx - 1.0f)) * 0.125f;
    float va = v[off], vb = v[off + 32];
    float mn = fmaxf(m, s);
    float corr = __expf(m - mn);
    float w = __expf(s - mn);
    Z  = Z  * corr + w;
    za = za * corr + w * va;
    zb = zb * corr + w * vb;
    m = mn;
  }
  const float inv = 1.0f / Z;
  z[qoff] = za * inv;
  z[qoff + 32] = zb * inv;
}

// ---------------------------------------------------------------------------
// Head output: out[t,o] = sum_d hh[t,d]*W2[j,d,o] + b2[j,o], o<3, j = t%17.
// ---------------------------------------------------------------------------
__global__ __launch_bounds__(256) void head_out_k(
    const float* __restrict__ hh, const float* __restrict__ W2,
    const float* __restrict__ b2, float* __restrict__ out, int ntok)
{
  int idx = blockIdx.x * 256 + threadIdx.x;
  if (idx >= ntok * 3) return;
  int t = idx / 3, o = idx % 3;
  int j = t % 17;
  const float* hr = hh + (size_t)t * 512;
  const float* wr = W2 + (size_t)j * 512 * 3 + o;
  float acc = b2[j * 3 + o];
  for (int d = 0; d < 512; ++d) acc += hr[d] * wr[d * 3];
  out[idx] = acc;
}

// ---------------------------------------------------------------------------
// Host side. Input order: x, x_vel, topo_bias, then params flattened as a
// JAX pytree (dict keys sorted alphabetically, lists/tuples in order).
// ---------------------------------------------------------------------------
extern "C" void kernel_launch(void* const* d_in, const int* in_sizes, int n_in,
                              void* d_out, int out_size, void* d_ws, size_t ws_size,
                              hipStream_t stream) {
  (void)in_sizes; (void)n_in; (void)out_size; (void)ws_size;

  const int B = 4, T = 243, J = 17, D = 512, NBT = B * T;
  const int TOK = B * T * J;                 // 16524
  const int WINDOWS[3] = {3, 9, 27};

  auto F = [&](int i) -> const float* { return (const float*)d_in[i]; };

  // --- input index map (pytree order) ---
  const int I_X = 0, I_XV = 1, I_A = 2;
  const int I_Wp = 3, I_eWvel = 4, I_bp = 5, I_bvel = 6;
  const int I_hW1 = 7, I_hW2 = 8, I_hb1 = 9, I_hb2 = 10, I_hlng = 11, I_hlnb = 12;
  const int I_jemb = 13;
  const int SP0 = 14;  // spatial block stride 17:
  // +0 Wk, +1 Wo, +2 Wq, +3 Wv, +4 Wvel, +5 bo, +6 gamma,
  // +7 fW1, +8 fW2, +9 fb1, +10 fb2, +11 ln1g, +12 ln1b, +13 ln2g, +14 ln2b,
  // +15 lnvg, +16 lnvb
  const int TP0 = 65;  // temporal block stride 13:
  // +0 Wk, +1 Wo, +2 Wq, +3 Wv, +4 bo, +5 fW1, +6 fW2, +7 fb1, +8 fb2,
  // +9 ln1g, +10 ln1b, +11 ln2g, +12 ln2b

  // --- workspace layout (floats) ---
  float* X   = (float*)d_ws;                 // hidden state
  float* Vv  = X   + (size_t)TOK * D;        // velocity embedding
  float* L1  = Vv  + (size_t)TOK * D;        // ln out / attn z (reused)
  float* L2  = L1  + (size_t)TOK * D;        // lnv out
  float* Qb  = L2  + (size_t)TOK * D;
  float* Kb  = Qb  + (size_t)TOK * D;
  float* Vb  = Kb  + (size_t)TOK * D;
  float* Hid = Vb  + (size_t)TOK * D;        // TOK*2048 ffn hidden / head hidden

  auto gemm = [&](const float* A, int lda, const float* Bm, int ldb,
                  const float* bias, const float* res,
                  float* C, int ldc, int M, int N, int K, int flags) {
    dim3 grid((M + TILE_M - 1) / TILE_M, N / TILE_N);
    gemm_f16wmma<<<grid, 256, 0, stream>>>(A, lda, Bm, ldb, bias, res,
                                           C, ldc, M, N, K, flags);
  };
  auto ln = [&](const float* x, const float* g, const float* b, float* y) {
    layernorm_k<<<TOK, 256, 0, stream>>>(x, g, b, y);
  };

  // --- embedding ---
  embed_k<<<TOK, 256, 0, stream>>>(F(I_X), F(I_XV), F(I_Wp), F(I_bp),
                                   F(I_eWvel), F(I_bvel), F(I_jemb), X, Vv);

  for (int blk = 0; blk < 3; ++blk) {
    // ================= spatial =================
    const int sp = SP0 + 17 * blk;
    ln(X,  F(sp + 11), F(sp + 12), L1);            // ln1
    ln(Vv, F(sp + 15), F(sp + 16), L2);            // lnv
    gemm(L1, D, F(sp + 2), D, nullptr, nullptr, Qb, D, TOK, D, D, 0);  // Wq
    gemm(L1, D, F(sp + 0), D, nullptr, nullptr, Kb, D, TOK, D, D, 0);  // Wk
    gemm(L1, D, F(sp + 3), D, nullptr, nullptr, Vb, D, TOK, D, D, 0);  // Wv
    gemm(L2, D, F(sp + 4), D, nullptr, Vb,      Vb, D, TOK, D, D, 2);  // +Wvel
    spatial_attn_k<<<dim3(NBT, 8), 256, 0, stream>>>(Qb, Kb, Vb, F(I_A),
                                                     F(sp + 6), L1);
    gemm(L1, D, F(sp + 1), D, F(sp + 5), X, X, D, TOK, D, D, 2);       // Wo + res
    ln(X, F(sp + 13), F(sp + 14), L1);             // ln2
    gemm(L1,  D,    F(sp + 7), 2048, F(sp + 9), nullptr, Hid, 2048,
         TOK, 2048, D, 1);                                             // FFN1+gelu
    gemm(Hid, 2048, F(sp + 8), D,    F(sp + 10), X, X, D,
         TOK, D, 2048, 2);                                             // FFN2+res

    // ================= temporal =================
    const int tp = TP0 + 13 * blk;
    ln(X, F(tp + 9), F(tp + 10), L1);              // ln1
    gemm(L1, D, F(tp + 2), D, nullptr, nullptr, Qb, D, TOK, D, D, 0);  // Wq
    gemm(L1, D, F(tp + 0), D, nullptr, nullptr, Kb, D, TOK, D, D, 0);  // Wk
    gemm(L1, D, F(tp + 3), D, nullptr, nullptr, Vb, D, TOK, D, D, 0);  // Wv
    temporal_attn_k<<<dim3((T + 7) / 8, B * J * 8), 256, 0, stream>>>(
        Qb, Kb, Vb, L1, B, T, J, WINDOWS[blk] / 2);
    gemm(L1, D, F(tp + 1), D, F(tp + 4), X, X, D, TOK, D, D, 2);       // Wo + res
    ln(X, F(tp + 11), F(tp + 12), L1);             // ln2
    gemm(L1,  D,    F(tp + 5), 2048, F(tp + 7), nullptr, Hid, 2048,
         TOK, 2048, D, 1);
    gemm(Hid, 2048, F(tp + 6), D,    F(tp + 8), X, X, D,
         TOK, D, 2048, 2);
  }

  // ================= head =================
  ln(X, F(I_hlng), F(I_hlnb), L1);
  // per-joint GEMM: rows = (b,t) with fixed joint j; strided views via lda/ldc
  for (int j = 0; j < J; ++j) {
    gemm(L1 + j * D, J * D,
         F(I_hW1) + (size_t)j * D * D, D,
         F(I_hb1) + j * D, nullptr,
         Hid + j * D, J * D,
         NBT, D, D, 1 /*gelu*/);
  }
  head_out_k<<<(TOK * 3 + 255) / 256, 256, 0, stream>>>(
      Hid, F(I_hW2), F(I_hb2), (float*)d_out, TOK);
}
